// GraphEdgeAttenNetwork_33663953666630
// MI455X (gfx1250) — matrix operand order repaired and checked by reading, not compiled, not run on hardware
//
#include <hip/hip_runtime.h>

// ---------------------------------------------------------------------------
// Types / helpers
// ---------------------------------------------------------------------------
typedef __attribute__((ext_vector_type(16))) __bf16 v16bf;
typedef __attribute__((ext_vector_type(8)))  __bf16 v8bf;
typedef __attribute__((ext_vector_type(8)))  float  v8f;

#define HS_LOG2 20
#define HSIZE   (1u << HS_LOG2)
#define HMASK   (HSIZE - 1u)

__device__ __forceinline__ unsigned short bf_bits(__bf16 b) {
  return __builtin_bit_cast(unsigned short, b);
}
__device__ __forceinline__ __bf16 bits_bf(unsigned short s) {
  return __builtin_bit_cast(__bf16, s);
}
__device__ __forceinline__ __bf16 f2bf(float f) {
  unsigned u = __builtin_bit_cast(unsigned, f);
  unsigned r = u + 0x7FFFu + ((u >> 16) & 1u);   // round-to-nearest-even
  return bits_bf((unsigned short)(r >> 16));
}
__device__ __forceinline__ float bf2f(__bf16 b) {
  unsigned u = ((unsigned)bf_bits(b)) << 16;
  return __builtin_bit_cast(float, u);
}

__device__ __forceinline__ unsigned hash64(unsigned long long k) {
  k *= 0x9E3779B97F4A7C15ull; k ^= k >> 29;
  k *= 0xBF58476D1CE4E5B9ull; k ^= k >> 32;
  return (unsigned)k;
}

// float atomic max via signed/unsigned bit-pattern split (init must be -inf)
__device__ __forceinline__ void atomicMaxF(float* addr, float val) {
  if (val >= 0.f) atomicMax((int*)addr, __builtin_bit_cast(int, val));
  else            atomicMin((unsigned*)addr, __builtin_bit_cast(unsigned, val));
}

// ---------------------------------------------------------------------------
// WMMA fragment helpers (CDNA5 16x16x32 bf16, wave32; layouts per ISA 7.12.2)
// ---------------------------------------------------------------------------
__device__ __forceinline__ v8f wmma_bf16(v16bf a, v16bf b, v8f c) {
  return __builtin_amdgcn_wmma_f32_16x16x32_bf16(
      /*neg_a=*/false, a, /*neg_b=*/false, b,
      /*c_mod=*/(short)0, c, /*reuse_a=*/false, /*reuse_b=*/false);
}

// A (16x32 bf16) from row-major LDS. lane(&15)=row M; elems 0..7 hold
// K = ksel*8 + i, elems 8..15 hold K = 16 + ksel*8 + (i-8). With ld, k0, ks
// all multiples of 8 elements, each half is one aligned 16-byte LDS load.
__device__ __forceinline__ v16bf load_A_lds(const __bf16* __restrict__ base,
                                            int ld, int k0) {
  int lane = threadIdx.x & 31;
  int m = lane & 15;
  int ks = (lane >> 4) * 8;
  const __bf16* row = base + (size_t)m * ld + k0 + ks;
  v8bf lo = *(const v8bf*)(row);        // ds_load_b128
  v8bf hi = *(const v8bf*)(row + 16);   // ds_load_b128
  return __builtin_shufflevector(lo, hi, 0, 1, 2, 3, 4, 5, 6, 7,
                                 8, 9, 10, 11, 12, 13, 14, 15);
}

// B fragment from fragment-packed weights: tile = nt*ktiles + kt, each lane's
// 16 bf16 stored contiguously -> two coalesced global_load_b128 per wave-lane.
__device__ __forceinline__ v16bf load_B_pk(const __bf16* __restrict__ Wp,
                                           int tile) {
  int lane = threadIdx.x & 31;
  const v8bf* ptr = (const v8bf*)(Wp + ((size_t)tile << 9) + (lane << 4));
  v8bf lo = ptr[0];
  v8bf hi = ptr[1];
  return __builtin_shufflevector(lo, hi, 0, 1, 2, 3, 4, 5, 6, 7,
                                 8, 9, 10, 11, 12, 13, 14, 15);
}

// One 16x16 output tile: A_lds[16 x kdim] @ packed W (K=kdim), output tile nt
__device__ __forceinline__ v8f tile_gemm(const __bf16* __restrict__ Alds,
                                         int lda, int kdim,
                                         const __bf16* __restrict__ Wp,
                                         int nt) {
  const int ktiles = kdim >> 5;
  v8f c;
#pragma unroll
  for (int i = 0; i < 8; ++i) c[i] = 0.f;
  for (int kt = 0; kt < ktiles; ++kt) {
    v16bf a = load_A_lds(Alds, lda, kt * 32);
    v16bf b = load_B_pk(Wp, nt * ktiles + kt);
    c = wmma_bf16(a, b, c);
  }
  return c;
}

// C layout: elem i -> M = i + 8*(lane>=16), N = lane&15
__device__ __forceinline__ void store_C_f32(float* base_n0, int ld, v8f c,
                                            const float* bias_n0, bool relu) {
  int lane = threadIdx.x & 31;
  int n = lane & 15;
  int mb = (lane >> 4) * 8;
#pragma unroll
  for (int i = 0; i < 8; ++i) {
    float v = c[i] + bias_n0[n];
    if (relu) v = fmaxf(v, 0.f);
    base_n0[(size_t)(mb + i) * ld + n] = v;
  }
}
__device__ __forceinline__ void store_C_bf16(__bf16* base_n0, int ld, v8f c,
                                             const float* bias_n0, bool relu) {
  int lane = threadIdx.x & 31;
  int n = lane & 15;
  int mb = (lane >> 4) * 8;
#pragma unroll
  for (int i = 0; i < 8; ++i) {
    float v = c[i] + bias_n0[n];
    if (relu) v = fmaxf(v, 0.f);
    base_n0[(size_t)(mb + i) * ld + n] = f2bf(v);
  }
}

// ---------------------------------------------------------------------------
// Init + weight fragment-packing + reverse-edge hash
// ---------------------------------------------------------------------------
__global__ void k_init(unsigned long long* hk, int* hv, float* xxm,
                       float* subj, float* obj, int* cs, int* cd,
                       int hs, int n128, int n) {
  int i = blockIdx.x * 256 + threadIdx.x;
  int stride = gridDim.x * 256;
  for (int j = i; j < hs; j += stride) { hk[j] = ~0ull; hv[j] = 0x7FFFFFFF; }
  for (int j = i; j < n128; j += stride) {
    xxm[j] = -__builtin_inff(); subj[j] = 0.f; obj[j] = 0.f;
  }
  for (int j = i; j < n; j += stride) { cs[j] = 0; cd[j] = 0; }
}

// Pack logical W[K][N] (bf16) into WMMA B-fragment order:
// dst[((nt*ktiles)+kt)*512 + lane*16 + elem], where
//   n = nt*16 + (lane&15), k = kt*32 + (lane>>4)*8 + (elem&7) + (elem>>3)*16.
// trans==0: W[k][n] = src[k*src_cols + n]   (src is [K][N], src_cols=N)
// trans==1: W[k][n] = src[n*src_cols + k]   (src is [N][K], src_cols=K)
__global__ void k_pack(const float* __restrict__ src, __bf16* __restrict__ dst,
                       int K, int N, int trans, int src_cols) {
  int i = blockIdx.x * 256 + threadIdx.x;
  if (i >= K * N) return;
  int elem = i & 15;
  int lane = (i >> 4) & 31;
  int tile = i >> 9;
  int ktiles = K >> 5;
  int nt = tile / ktiles;
  int kt = tile - nt * ktiles;
  int n = nt * 16 + (lane & 15);
  int k = kt * 32 + (lane >> 4) * 8 + (elem & 7) + (elem >> 3) * 16;
  float v = trans ? src[(size_t)n * src_cols + k]
                  : src[(size_t)k * src_cols + n];
  dst[i] = f2bf(v);
}

__global__ void k_hins(const int* __restrict__ ei, unsigned long long* hk,
                       int* hv, int E, int N) {
  int e = blockIdx.x * 256 + threadIdx.x;
  if (e >= E) return;
  unsigned long long key =
      (unsigned long long)(unsigned)ei[e] * (unsigned)N + (unsigned)ei[E + e];
  unsigned slot = hash64(key) & HMASK;
  for (;;) {
    unsigned long long prev = atomicCAS(&hk[slot], ~0ull, key);
    if (prev == ~0ull || prev == key) { atomicMin(&hv[slot], e); break; }
    slot = (slot + 1) & HMASK;
  }
}

__global__ void k_hprobe(const int* __restrict__ ei,
                         const unsigned long long* __restrict__ hk,
                         const int* __restrict__ hv, int* __restrict__ rev,
                         int E, int N) {
  int e = blockIdx.x * 256 + threadIdx.x;
  if (e >= E) return;
  unsigned long long rkey =
      (unsigned long long)(unsigned)ei[E + e] * (unsigned)N + (unsigned)ei[e];
  unsigned slot = hash64(rkey) & HMASK;
  int r = -1;
  for (;;) {
    unsigned long long k = hk[slot];
    if (k == ~0ull) break;
    if (k == rkey) { r = hv[slot]; break; }
    slot = (slot + 1) & HMASK;
  }
  rev[e] = r;
}

// ---------------------------------------------------------------------------
// Fused edge kernel: one wave == one 16-edge tile
// ---------------------------------------------------------------------------
struct EdgeP {
  const float* x; const float* ef; const int* ei; const int* rev;
  const __bf16 *Wg1b, *We1b, *We2b, *Wqb, *Wpeb, *Wvb, *A1Tb, *A2Tb;
  const float *bg1, *Wg2, *bg2, *be1, *be2, *ge, *beta_e, *bq, *bpe, *bv, *a1, *a2;
  float* out_gcn; float *subj, *obj, *xxmax; int *cnt_s, *cnt_d;
  int N, E;
};

__global__ __launch_bounds__(32) void k_edge(EdgeP p) {
  __shared__ alignas(16) __bf16 sIn[16][512];   // [x_i | ef | gated rev | x_j]
  __shared__ alignas(16) __bf16 sQEV[16][384];  // MLP hidden, later q|e|v
  __shared__ alignas(16) float  sF[16][128];    // gcn pre-LN, later xx_e
  __shared__ alignas(16) float  sAux[16][64];   // gate hidden, later logits
  __shared__ alignas(16) __bf16 sHa[16][64];    // attention head input
  __shared__ alignas(16) __bf16 sHb[16][64];    // attention hidden (relu)
  __shared__ float sGate[16];
  __shared__ int   sSrc[16], sDst[16];

  const int lane = threadIdx.x;
  const int e0 = blockIdx.x * 16;
  const int E = p.E;

  if (lane < 16) {
    int e = min(e0 + lane, E - 1);
    sSrc[lane] = p.ei[e];
    sDst[lane] = p.ei[E + e];
  }
  __syncthreads();

  // gather & concat (bf16)
  for (int t = lane; t < 16 * 128; t += 32) {
    int r = t >> 7, c = t & 127;
    int e = min(e0 + r, E - 1);
    sIn[r][c]       = f2bf(p.x[(size_t)sSrc[r] * 128 + c]);
    sIn[r][128 + c] = f2bf(p.ef[(size_t)e * 128 + c]);
    int rv = p.rev[e];
    sIn[r][256 + c] = f2bf(rv >= 0 ? p.ef[(size_t)rv * 128 + c] : 0.f);
    sIn[r][384 + c] = f2bf(p.x[(size_t)sDst[r] * 128 + c]);
  }
  __syncthreads();

  // ---- edge gate: relu(ef@Wg1+bg1) @ Wg2 + bg2 -> sigmoid ----
  for (int nt = 0; nt < 4; ++nt) {
    v8f c = tile_gemm(&sIn[0][128], 512, 128, p.Wg1b, nt);
    store_C_f32(&sAux[0][nt * 16], 64, c, p.bg1 + nt * 16, true);
  }
  __syncthreads();
  if (lane < 16) {
    float s = p.bg2[0];
    for (int j = 0; j < 64; ++j) s += sAux[lane][j] * p.Wg2[j];
    sGate[lane] = 1.f / (1.f + __expf(-s));
  }
  __syncthreads();
  for (int t = lane; t < 16 * 128; t += 32) {
    int r = t >> 7, c = t & 127;
    sIn[r][256 + c] = f2bf(bf2f(sIn[r][256 + c]) * sGate[r]);
  }
  __syncthreads();

  // ---- nn_edge MLP: [16,512]@We1 -> relu -> @We2 ----
#pragma unroll 1
  for (int nt = 0; nt < 24; ++nt) {
    v8f c = tile_gemm(&sIn[0][0], 512, 512, p.We1b, nt);
    store_C_bf16(&sQEV[0][nt * 16], 384, c, p.be1 + nt * 16, true);
  }
  __syncthreads();
#pragma unroll 1
  for (int nt = 0; nt < 8; ++nt) {
    v8f c = tile_gemm(&sQEV[0][0], 384, 384, p.We2b, nt);
    store_C_f32(&sF[0][nt * 16], 128, c, p.be2 + nt * 16, false);
  }
  __syncthreads();

  // ---- LayerNorm -> gcn_edge_feature (output 2) ----
  if (lane < 16) {
    int e = e0 + lane;
    float m = 0.f;
    for (int c = 0; c < 128; ++c) m += sF[lane][c];
    m *= (1.f / 128.f);
    float v = 0.f;
    for (int c = 0; c < 128; ++c) { float d = sF[lane][c] - m; v += d * d; }
    v *= (1.f / 128.f);
    float inv = rsqrtf(v + 1e-5f);
    for (int c = 0; c < 128; ++c) {
      float g = (sF[lane][c] - m) * inv * p.ge[c] + p.beta_e[c];
      sF[lane][c] = g;
      if (e < E) p.out_gcn[(size_t)e * 128 + c] = g;
    }
  }
  __syncthreads();

  // segment sums / counts for the twinning means
  for (int t = lane; t < 16 * 128; t += 32) {
    int r = t >> 7, c = t & 127;
    if (e0 + r < E) {
      atomicAdd(&p.subj[(size_t)sSrc[r] * 128 + c], sF[r][c]);
      atomicAdd(&p.obj[(size_t)sDst[r] * 128 + c], sF[r][c]);
    }
  }
  if (lane < 16 && e0 + lane < E) {
    atomicAdd(&p.cnt_s[sSrc[lane]], 1);
    atomicAdd(&p.cnt_d[sDst[lane]], 1);
  }
  __syncthreads();

  // ---- q / e / v projections ----
#pragma unroll 1
  for (int nt = 0; nt < 8; ++nt) {
    v8f cq = tile_gemm(&sIn[0][0], 512, 128, p.Wqb, nt);
    store_C_bf16(&sQEV[0][nt * 16], 384, cq, p.bq + nt * 16, false);
    v8f ce = tile_gemm(&sIn[0][128], 512, 128, p.Wpeb, nt);
    store_C_bf16(&sQEV[0][128 + nt * 16], 384, ce, p.bpe + nt * 16, false);
    v8f cv = tile_gemm(&sIn[0][384], 512, 128, p.Wvb, nt);
    store_C_bf16(&sQEV[0][256 + nt * 16], 384, cv, p.bv + nt * 16, false);
  }
  __syncthreads();

  // ---- per-head attention MLP + softmax + prob*value ----
#pragma unroll 1
  for (int h = 0; h < 4; ++h) {
    // head channels are strided by H=4: c' -> flat c'*4 + h
    for (int t = lane; t < 16 * 64; t += 32) {
      int r = t >> 6, cc = t & 63;
      sHa[r][cc] = (cc < 32) ? sQEV[r][cc * 4 + h]
                             : sQEV[r][128 + (cc - 32) * 4 + h];
    }
    __syncthreads();
    for (int nt = 0; nt < 4; ++nt) {
      v8f c = tile_gemm(&sHa[0][0], 64, 64, p.A1Tb, nt);
      store_C_bf16(&sHb[0][nt * 16], 64, c, p.a1 + nt * 16, true);
    }
    __syncthreads();
    for (int nt = 0; nt < 2; ++nt) {
      v8f c = tile_gemm(&sHb[0][0], 64, 64, p.A2Tb, nt);
      store_C_f32(&sAux[0][nt * 16], 64, c, p.a2 + nt * 16, false);
    }
    __syncthreads();
    if (lane < 16) {
      float mx = -1e30f;
      for (int o = 0; o < 32; ++o) mx = fmaxf(mx, sAux[lane][o]);
      float s = 0.f;
      for (int o = 0; o < 32; ++o) {
        float pr = __expf(sAux[lane][o] - mx);
        sAux[lane][o] = pr; s += pr;
      }
      float inv = 1.f / s;
      for (int o = 0; o < 32; ++o) {
        float prob = sAux[lane][o] * inv;
        sF[lane][o * 4 + h] = prob * bf2f(sQEV[lane][256 + o * 4 + h]);
      }
    }
    __syncthreads();
  }

  // scatter-max into xxmax[src]
  for (int t = lane; t < 16 * 128; t += 32) {
    int r = t >> 7, c = t & 127;
    if (e0 + r < E) atomicMaxF(&p.xxmax[(size_t)sSrc[r] * 128 + c], sF[r][c]);
  }
}

// ---------------------------------------------------------------------------
// Fused node kernel: one wave == one 16-node tile
// ---------------------------------------------------------------------------
struct NodeP {
  const float* x;
  const __bf16 *Wtb, *Wp1b, *Wp2b;
  const float *bt, *bp1, *bp2, *gn, *bn;
  const float *subj, *obj, *xxmax; const int *cnt_s, *cnt_d;
  float* out_xx;
  int N;
};

__global__ __launch_bounds__(32) void k_node(NodeP p) {
  __shared__ alignas(16) __bf16 sCat[16][256];  // [subj|obj], later [x|xx]
  __shared__ alignas(16) __bf16 sH[16][256];    // prop hidden
  __shared__ alignas(16) float  sF[16][128];    // twin result / prop pre-LN
  const int lane = threadIdx.x;
  const int n0 = blockIdx.x * 16;
  const int Nn = p.N;

  for (int t = lane; t < 16 * 128; t += 32) {
    int r = t >> 7, c = t & 127;
    int nd = min(n0 + r, Nn - 1);
    float cs = fmaxf((float)p.cnt_s[nd], 1.f);
    float cd = fmaxf((float)p.cnt_d[nd], 1.f);
    sCat[r][c]       = f2bf(p.subj[(size_t)nd * 128 + c] / cs);
    sCat[r][128 + c] = f2bf(p.obj[(size_t)nd * 128 + c] / cd);
  }
  __syncthreads();

  // twinning linear: [16,256]@Wt + bt
#pragma unroll 1
  for (int nt = 0; nt < 8; ++nt) {
    v8f c = tile_gemm(&sCat[0][0], 256, 256, p.Wtb, nt);
    store_C_f32(&sF[0][nt * 16], 128, c, p.bt + nt * 16, false);
  }
  __syncthreads();

  // xx = relu(segmax) * sigmoid(twin); build [x | xx]
  for (int t = lane; t < 16 * 128; t += 32) {
    int r = t >> 7, c = t & 127;
    int nd = min(n0 + r, Nn - 1);
    float xr = fmaxf(p.xxmax[(size_t)nd * 128 + c], 0.f); // -inf init -> 0
    float sg = 1.f / (1.f + __expf(-sF[r][c]));
    sCat[r][c]       = f2bf(p.x[(size_t)nd * 128 + c]);
    sCat[r][128 + c] = f2bf(xr * sg);
  }
  __syncthreads();

  // prop MLP: [16,256]@Wp1 -> relu -> @Wp2
#pragma unroll 1
  for (int nt = 0; nt < 16; ++nt) {
    v8f c = tile_gemm(&sCat[0][0], 256, 256, p.Wp1b, nt);
    store_C_bf16(&sH[0][nt * 16], 256, c, p.bp1 + nt * 16, true);
  }
  __syncthreads();
#pragma unroll 1
  for (int nt = 0; nt < 8; ++nt) {
    v8f c = tile_gemm(&sH[0][0], 256, 256, p.Wp2b, nt);
    store_C_f32(&sF[0][nt * 16], 128, c, p.bp2 + nt * 16, false);
  }
  __syncthreads();

  // final LayerNorm -> out xx
  if (lane < 16) {
    int nd = n0 + lane;
    if (nd < Nn) {
      float m = 0.f;
      for (int c = 0; c < 128; ++c) m += sF[lane][c];
      m *= (1.f / 128.f);
      float v = 0.f;
      for (int c = 0; c < 128; ++c) { float d = sF[lane][c] - m; v += d * d; }
      v *= (1.f / 128.f);
      float inv = rsqrtf(v + 1e-5f);
      for (int c = 0; c < 128; ++c)
        p.out_xx[(size_t)nd * 128 + c] =
            (sF[lane][c] - m) * inv * p.gn[c] + p.bn[c];
    }
  }
}

// ---------------------------------------------------------------------------
// Host launch
// ---------------------------------------------------------------------------
extern "C" void kernel_launch(void* const* d_in, const int* in_sizes, int n_in,
                              void* d_out, int out_size, void* d_ws,
                              size_t ws_size, hipStream_t stream) {
  (void)n_in; (void)out_size; (void)ws_size;
  const float* x      = (const float*)d_in[0];
  const float* ef     = (const float*)d_in[1];
  const int*   ei     = (const int*)d_in[2];
  const float* Wg1    = (const float*)d_in[3];
  const float* bg1    = (const float*)d_in[4];
  const float* Wg2    = (const float*)d_in[5];
  const float* bg2    = (const float*)d_in[6];
  const float* We1    = (const float*)d_in[7];
  const float* be1    = (const float*)d_in[8];
  const float* We2    = (const float*)d_in[9];
  const float* be2    = (const float*)d_in[10];
  const float* ge     = (const float*)d_in[11];
  const float* beta_e = (const float*)d_in[12];
  const float* Wq     = (const float*)d_in[13];
  const float* bq     = (const float*)d_in[14];
  const float* Wpe    = (const float*)d_in[15];
  const float* bpe    = (const float*)d_in[16];
  const float* Wv     = (const float*)d_in[17];
  const float* bv     = (const float*)d_in[18];
  const float* A1     = (const float*)d_in[19];
  const float* a1     = (const float*)d_in[20];
  const float* A2     = (const float*)d_in[21];
  const float* a2     = (const float*)d_in[22];
  const float* Wt     = (const float*)d_in[23];
  const float* bt     = (const float*)d_in[24];
  const float* Wp1    = (const float*)d_in[25];
  const float* bp1    = (const float*)d_in[26];
  const float* Wp2    = (const float*)d_in[27];
  const float* bp2    = (const float*)d_in[28];
  const float* gn     = (const float*)d_in[29];
  const float* bn     = (const float*)d_in[30];

  const int N = in_sizes[0] / 128;
  const int E = in_sizes[2] / 2;

  float* out_xx  = (float*)d_out;
  float* out_gcn = out_xx + (size_t)N * 128;

  // workspace carve-up
  char* w = (char*)d_ws;
  size_t off = 0;
  auto take = [&](size_t bytes) -> void* {
    void* ptr = w + off;
    off = (off + bytes + 255) & ~(size_t)255;
    return ptr;
  };
  unsigned long long* hk = (unsigned long long*)take((size_t)HSIZE * 8);
  int*    hv    = (int*)take((size_t)HSIZE * 4);
  int*    rev   = (int*)take((size_t)E * 4);
  float*  xxmax = (float*)take((size_t)N * 128 * 4);
  float*  subj  = (float*)take((size_t)N * 128 * 4);
  float*  obj   = (float*)take((size_t)N * 128 * 4);
  int*    cs    = (int*)take((size_t)N * 4);
  int*    cd    = (int*)take((size_t)N * 4);
  __bf16* Wg1b  = (__bf16*)take((size_t)128 * 64 * 2);
  __bf16* We1b  = (__bf16*)take((size_t)512 * 384 * 2);
  __bf16* We2b  = (__bf16*)take((size_t)384 * 128 * 2);
  __bf16* Wqb   = (__bf16*)take((size_t)128 * 128 * 2);
  __bf16* Wpeb  = (__bf16*)take((size_t)128 * 128 * 2);
  __bf16* Wvb   = (__bf16*)take((size_t)128 * 128 * 2);
  __bf16* A1Tb  = (__bf16*)take((size_t)64 * 64 * 2);
  __bf16* A2Tb  = (__bf16*)take((size_t)64 * 32 * 2);
  __bf16* Wtb   = (__bf16*)take((size_t)256 * 128 * 2);
  __bf16* Wp1b  = (__bf16*)take((size_t)256 * 256 * 2);
  __bf16* Wp2b  = (__bf16*)take((size_t)256 * 256 * 2);

  // 1) init accumulators + hash table
  k_init<<<2048, 256, 0, stream>>>(hk, hv, xxmax, subj, obj, cs, cd,
                                   (int)HSIZE, N * 128, N);

  // 2) weight fragment-packing (bf16, WMMA B layout; A1/A2 transposed)
  auto pack = [&](const float* s, __bf16* d, int K, int Nn, int tr, int sc) {
    k_pack<<<(K * Nn + 255) / 256, 256, 0, stream>>>(s, d, K, Nn, tr, sc);
  };
  pack(Wg1, Wg1b, 128, 64, 0, 64);
  pack(We1, We1b, 512, 384, 0, 384);
  pack(We2, We2b, 384, 128, 0, 128);
  pack(Wq,  Wqb,  128, 128, 0, 128);
  pack(Wpe, Wpeb, 128, 128, 0, 128);
  pack(Wv,  Wvb,  128, 128, 0, 128);
  pack(A1,  A1Tb, 64, 64, 1, 64);   // W[k=c][n=o] = A1[o][c]
  pack(A2,  A2Tb, 64, 32, 1, 64);   // W[k=c][n=o] = A2[o][c]
  pack(Wt,  Wtb,  256, 128, 0, 128);
  pack(Wp1, Wp1b, 256, 256, 0, 256);
  pack(Wp2, Wp2b, 256, 128, 0, 128);

  // 3) reverse-edge lookup
  k_hins<<<(E + 255) / 256, 256, 0, stream>>>(ei, hk, hv, E, N);
  k_hprobe<<<(E + 255) / 256, 256, 0, stream>>>(ei, hk, hv, rev, E, N);

  // 4) fused edge kernel
  EdgeP ep;
  ep.x = x; ep.ef = ef; ep.ei = ei; ep.rev = rev;
  ep.Wg1b = Wg1b; ep.We1b = We1b; ep.We2b = We2b;
  ep.Wqb = Wqb; ep.Wpeb = Wpeb; ep.Wvb = Wvb; ep.A1Tb = A1Tb; ep.A2Tb = A2Tb;
  ep.bg1 = bg1; ep.Wg2 = Wg2; ep.bg2 = bg2; ep.be1 = be1; ep.be2 = be2;
  ep.ge = ge; ep.beta_e = beta_e; ep.bq = bq; ep.bpe = bpe; ep.bv = bv;
  ep.a1 = a1; ep.a2 = a2;
  ep.out_gcn = out_gcn; ep.subj = subj; ep.obj = obj; ep.xxmax = xxmax;
  ep.cnt_s = cs; ep.cnt_d = cd; ep.N = N; ep.E = E;
  k_edge<<<(E + 15) / 16, 32, 0, stream>>>(ep);

  // 5) fused node kernel
  NodeP np;
  np.x = x; np.Wtb = Wtb; np.Wp1b = Wp1b; np.Wp2b = Wp2b;
  np.bt = bt; np.bp1 = bp1; np.bp2 = bp2; np.gn = gn; np.bn = bn;
  np.subj = subj; np.obj = obj; np.xxmax = xxmax;
  np.cnt_s = cs; np.cnt_d = cd; np.out_xx = out_xx; np.N = N;
  k_node<<<(N + 15) / 16, 32, 0, stream>>>(np);
}
